// SINE_71880572665939
// MI455X (gfx1250) — matrix operand order repaired and checked by reading, not compiled, not run on hardware
//
#include <hip/hip_runtime.h>
#include <hip/hip_bf16.h>
#include <math.h>

typedef __attribute__((ext_vector_type(16))) _Float16 v16h;
typedef __attribute__((ext_vector_type(8)))  _Float16 v8h;
typedef __attribute__((ext_vector_type(8)))  float    v8f;

#define B_TOT 4096
#define S_LEN 50
#define S_PAD 64
#define E_DIM 64
#define H_DIM 256
#define C_DIM 1000
#define K_TOP 4
#define NNEG  10
#define XLD   72      // padded LDS row stride (halfs): 144B, 16B-aligned, bank-friendly
#define NEG_INF_F (-1000000000.0f)

struct SH {
  alignas(16) _Float16 Xh[S_PAD * XLD];   // x_u (then x_u_hat) f16, rows [50,64) zero
  alignas(16) _Float16 Wt[128 * XLD];     // weight half-panel, TRANSPOSED: Wt[n][k]
  float    XY[S_PAD * E_DIM];             // xw3, later x_u_hat fp32. Aliased: s_u/argmax
  float    w2s[H_DIM];
  float    w5s[H_DIM];
  float    wk2s[H_DIM * K_TOP];
  float    hist_logit[S_PAD];
  float    att_logit[S_PAD];
  float    ck_logit[S_PAD * K_TOP];
  float    a_hist[S_PAD];
  float    attn_s[S_PAD];
  float    maskBias[S_PAD];
  float    a_ck[S_PAD * K_TOP];           // [s*4+k]
  float    c_u[K_TOP * E_DIM];
  float    c_un[K_TOP * E_DIM];
  float    phi[K_TOP * E_DIM];
  float    pl[K_TOP * S_PAD];
  float    p_us[K_TOP * S_PAD];
  float    zu[E_DIM];
  float    capt[E_DIM];
  float    vu[E_DIM];
  float    rinv[S_PAD];
  float    top_val[K_TOP];
  int      top_idx[K_TOP];
  float    nrm[K_TOP];
  float    ek[K_TOP];
  float    eu[K_TOP];
  float    scal;
};

__device__ __forceinline__ float fast_rcp(float x)  { return __builtin_amdgcn_rcpf(x); }
__device__ __forceinline__ float fast_tanh(float x) {
  float t = __expf(-2.0f * fabsf(x));           // branch-free v_exp_f32
  float r = (1.0f - t) * fast_rcp(1.0f + t);
  return copysignf(r, x);
}
__device__ __forceinline__ float fast_sigmoid(float x) {
  return fast_rcp(1.0f + __expf(-x));
}

// A fragment 16x32 f16 from row-major Xh (stride XLD):
// lane: elements 0..15 = [m0+hl][k0+hi*16 .. +16)   -> two ds_load_b128
__device__ __forceinline__ v16h load_A(const _Float16* X, int m0, int k0, int lane) {
  int hl = lane & 15, hi = lane >> 4;
  const _Float16* p = X + (m0 + hl) * XLD + k0 + hi * 16;
  v8h lo = *(const v8h*)p;
  v8h hv = *(const v8h*)(p + 8);
  return __builtin_shufflevector(lo, hv, 0,1,2,3,4,5,6,7,8,9,10,11,12,13,14,15);
}

// B fragment 32x16 f16 from TRANSPOSED Wt[n][k] (stride XLD):
// lane: elements 0..15 = Wt[n0+hl][k0+hi*16 .. +16)   -> two ds_load_b128
__device__ __forceinline__ v16h load_B(const _Float16* Wt, int k0, int n0, int lane) {
  int hl = lane & 15, hi = lane >> 4;
  const _Float16* p = Wt + (n0 + hl) * XLD + k0 + hi * 16;
  v8h lo = *(const v8h*)p;
  v8h hv = *(const v8h*)(p + 8);
  return __builtin_shufflevector(lo, hv, 0,1,2,3,4,5,6,7,8,9,10,11,12,13,14,15);
}

__global__ __launch_bounds__(256) void sine_fused_kernel(
    const int* __restrict__ hist_item, const int* __restrict__ pos_item,
    const int* __restrict__ neg_items, const float* __restrict__ item_emb,
    const float* __restrict__ concept_emb, const float* __restrict__ pos_emb,
    const float* __restrict__ w1, const float* __restrict__ w2,
    const float* __restrict__ w3, const float* __restrict__ wk1,
    const float* __restrict__ wk2, const float* __restrict__ w4,
    const float* __restrict__ w5, float* __restrict__ out) {
  __shared__ SH sh;
  const int b    = blockIdx.x;
  const int tid  = threadIdx.x;
  const int wave = tid >> 5;
  const int lane = tid & 31;
  const int hl   = lane & 15;
  const int hi   = lane >> 4;

  // ---------------- Phase 1: gather x_u, stage small weights, zero logits ----
  for (int i = tid; i < S_PAD * E_DIM; i += 256) {
    int s = i >> 6, e = i & 63;
    float v = 0.0f;
    if (s < S_LEN) {
      int it = hist_item[b * S_LEN + s];
      v = item_emb[(size_t)it * E_DIM + e] + pos_emb[s * E_DIM + e];
    }
    sh.Xh[s * XLD + e] = (_Float16)v;
  }
  for (int i = tid; i < H_DIM; i += 256) { sh.w2s[i] = w2[i]; sh.w5s[i] = w5[i]; }
  for (int i = tid; i < H_DIM * K_TOP; i += 256) sh.wk2s[i] = wk2[i];
  if (tid < S_PAD) {
    sh.hist_logit[tid] = 0.0f;
    sh.att_logit[tid]  = 0.0f;
    sh.maskBias[tid] = (tid < S_LEN && hist_item[b * S_LEN + tid] > 0) ? 0.0f : NEG_INF_F;
  }
  for (int i = tid; i < S_PAD * K_TOP; i += 256) sh.ck_logit[i] = 0.0f;
  __syncthreads();

  // ---------------- Phase 2: GEMM1 fused  tanh(x@w1)@w2 -> hist_logit --------
  for (int half = 0; half < 2; ++half) {
    for (int i = tid; i < 128 * E_DIM; i += 256) {
      int h = i & 127, e = i >> 7;                 // coalesced global read over h
      sh.Wt[h * XLD + e] = (_Float16)w1[e * H_DIM + half * 128 + h];
    }
    __syncthreads();
    for (int t = wave; t < 32; t += 8) {
      int m0 = (t & 3) * 16, n0 = (t >> 2) * 16;
      v8f c = {};
#pragma unroll
      for (int k0 = 0; k0 < E_DIM; k0 += 32) {
        v16h a = load_A(sh.Xh, m0, k0, lane);
        v16h bm = load_B(sh.Wt, k0, n0, lane);
        c = __builtin_amdgcn_wmma_f32_16x16x32_f16(false, a, false, bm, (short)0, c, false, false);
      }
      int n = half * 128 + n0 + hl;
      float w2v = sh.w2s[n];
#pragma unroll
      for (int i = 0; i < 8; ++i) {
        int row = m0 + hi * 8 + i;
        atomicAdd(&sh.hist_logit[row], fast_tanh(c[i]) * w2v);
      }
    }
    __syncthreads();
  }

  // softmax over s (masked) -> a_hist ; z_u
  if (tid == 0) {
    float mx = -3.0e38f;
    for (int s = 0; s < S_LEN; ++s) {
      float l = sh.hist_logit[s] + sh.maskBias[s];
      sh.a_hist[s] = l; mx = fmaxf(mx, l);
    }
    float sum = 0.0f;
    for (int s = 0; s < S_LEN; ++s) { float e = __expf(sh.a_hist[s] - mx); sh.a_hist[s] = e; sum += e; }
    float inv = 1.0f / sum;
    for (int s = 0; s < S_LEN; ++s) sh.a_hist[s] *= inv;
  }
  __syncthreads();
  if (tid < E_DIM) {
    float acc = 0.0f;
    for (int s = 0; s < S_LEN; ++s) acc += sh.a_hist[s] * (float)sh.Xh[s * XLD + tid];
    sh.zu[tid] = acc;
  }
  __syncthreads();

  // ---------------- Phase 3: GEMMk fused  tanh(x@wk1)@wk2 -> ck_logit --------
  for (int half = 0; half < 2; ++half) {
    for (int i = tid; i < 128 * E_DIM; i += 256) {
      int h = i & 127, e = i >> 7;
      sh.Wt[h * XLD + e] = (_Float16)wk1[e * H_DIM + half * 128 + h];
    }
    __syncthreads();
    for (int t = wave; t < 32; t += 8) {
      int m0 = (t & 3) * 16, n0 = (t >> 2) * 16;
      v8f c = {};
#pragma unroll
      for (int k0 = 0; k0 < E_DIM; k0 += 32) {
        v16h a = load_A(sh.Xh, m0, k0, lane);
        v16h bm = load_B(sh.Wt, k0, n0, lane);
        c = __builtin_amdgcn_wmma_f32_16x16x32_f16(false, a, false, bm, (short)0, c, false, false);
      }
      int n = half * 128 + n0 + hl;
#pragma unroll
      for (int i = 0; i < 8; ++i) {
        int row = m0 + hi * 8 + i;
        float tv = fast_tanh(c[i]);
#pragma unroll
        for (int k = 0; k < K_TOP; ++k)
          atomicAdd(&sh.ck_logit[row * K_TOP + k], tv * sh.wk2s[n * K_TOP + k]);
      }
    }
    __syncthreads();
  }
  if (tid < K_TOP) {  // softmax over s per k -> a_ck
    int k = tid;
    float mx = -3.0e38f;
    for (int s = 0; s < S_LEN; ++s) mx = fmaxf(mx, sh.ck_logit[s * K_TOP + k] + sh.maskBias[s]);
    float sum = 0.0f;
    for (int s = 0; s < S_LEN; ++s) {
      float e = __expf(sh.ck_logit[s * K_TOP + k] + sh.maskBias[s] - mx);
      sh.a_ck[s * K_TOP + k] = e; sum += e;
    }
    float inv = 1.0f / sum;
    for (int s = 0; s < S_LEN; ++s) sh.a_ck[s * K_TOP + k] *= inv;
  }

  // ---------------- Phase 4: s_u = z_u . concept_emb ; top-k(4) --------------
  float* su   = sh.XY;            // alias (XY written first in Phase 5)
  float* redV = sh.XY + 1024;
  int*   redI = (int*)(sh.XY + 1280);
  for (int c = 1000 + tid; c < 1024; c += 256) su[c] = -3.0e38f;
  __syncthreads();
  for (int c = tid; c < C_DIM; c += 256) {
    if (c + 256 < C_DIM) __builtin_prefetch(concept_emb + (size_t)(c + 256) * E_DIM, 0, 0);
    const float* ce = concept_emb + (size_t)c * E_DIM;
    float acc = 0.0f;
#pragma unroll 8
    for (int e = 0; e < E_DIM; ++e) acc += sh.zu[e] * ce[e];
    su[c] = acc;
  }
  __syncthreads();
  for (int r = 0; r < K_TOP; ++r) {
    float bv = -3.0e38f; int bi = 0;
    for (int c = tid; c < C_DIM; c += 256) { float v = su[c]; if (v > bv) { bv = v; bi = c; } }
    redV[tid] = bv; redI[tid] = bi;
    __syncthreads();
    for (int st = 128; st > 0; st >>= 1) {
      if (tid < st && redV[tid + st] > redV[tid]) { redV[tid] = redV[tid + st]; redI[tid] = redI[tid + st]; }
      __syncthreads();
    }
    if (tid == 0) { sh.top_val[r] = redV[0]; sh.top_idx[r] = redI[0]; su[redI[0]] = -3.0e38f; }
    __syncthreads();
  }
  {  // c_u + row norms + c_un
    int k = tid >> 6, e = tid & 63;
    float v = fast_sigmoid(sh.top_val[k]) * concept_emb[(size_t)sh.top_idx[k] * E_DIM + e];
    sh.c_u[k * E_DIM + e] = v;
  }
  __syncthreads();
  if (tid < K_TOP) {
    float ss = 0.0f;
    for (int e = 0; e < E_DIM; ++e) { float v = sh.c_u[tid * E_DIM + e]; ss += v * v; }
    sh.nrm[tid] = 1.0f / fmaxf(sqrtf(ss), 1e-12f);
  }
  __syncthreads();
  sh.c_un[tid] = sh.c_u[tid] * sh.nrm[tid >> 6];
  __syncthreads();

  // ---------------- Phase 5: xw3 = x @ w3 (WMMA), p_u ------------------------
  for (int i = tid; i < E_DIM * E_DIM; i += 256) {
    int n = i & 63, e = i >> 6;                   // transposed stage: Wt[n][e]
    sh.Wt[n * XLD + e] = (_Float16)w3[e * E_DIM + n];
  }
  __syncthreads();
  for (int t = wave; t < 16; t += 8) {
    int m0 = (t & 3) * 16, n0 = (t >> 2) * 16;
    v8f c = {};
#pragma unroll
    for (int k0 = 0; k0 < E_DIM; k0 += 32) {
      v16h a = load_A(sh.Xh, m0, k0, lane);
      v16h bm = load_B(sh.Wt, k0, n0, lane);
      c = __builtin_amdgcn_wmma_f32_16x16x32_f16(false, a, false, bm, (short)0, c, false, false);
    }
#pragma unroll
    for (int i = 0; i < 8; ++i) sh.XY[(m0 + hi * 8 + i) * E_DIM + n0 + hl] = c[i];
  }
  __syncthreads();
  if (tid < S_LEN) {
    float ss = 0.0f;
    for (int e = 0; e < E_DIM; ++e) { float v = sh.XY[tid * E_DIM + e]; ss += v * v; }
    sh.rinv[tid] = 1.0f / fmaxf(sqrtf(ss), 1e-12f);
  }
  __syncthreads();
  if (tid < K_TOP * S_LEN) {
    int k = tid / S_LEN, s = tid % S_LEN;
    float acc = 0.0f;
    for (int e = 0; e < E_DIM; ++e) acc += sh.XY[s * E_DIM + e] * sh.c_un[k * E_DIM + e];
    sh.pl[k * S_PAD + s] = acc * sh.rinv[s];
  }
  __syncthreads();
  if (tid < S_LEN) {  // softmax over k per s
    float mx = -3.0e38f;
    for (int k = 0; k < K_TOP; ++k) mx = fmaxf(mx, sh.pl[k * S_PAD + tid]);
    float sum = 0.0f;
    for (int k = 0; k < K_TOP; ++k) { float e = __expf(sh.pl[k * S_PAD + tid] - mx); sh.p_us[k * S_PAD + tid] = e; sum += e; }
    float inv = fast_rcp(sum);
    for (int k = 0; k < K_TOP; ++k) sh.p_us[k * S_PAD + tid] *= inv;
  }
  __syncthreads();

  // ---------------- Phase 6: phi_u then x_u_hat ------------------------------
  {
    int k = tid >> 6, e = tid & 63;
    float acc = 0.0f;
    for (int s = 0; s < S_LEN; ++s)
      acc += sh.p_us[k * S_PAD + s] * sh.a_ck[s * K_TOP + k] * (float)sh.Xh[s * XLD + e];
    sh.phi[k * E_DIM + e] = acc;
  }
  __syncthreads();
  for (int i = tid; i < S_LEN * E_DIM; i += 256) {
    int s = i >> 6, e = i & 63;
    float acc = 0.0f;
#pragma unroll
    for (int k = 0; k < K_TOP; ++k) acc += sh.p_us[k * S_PAD + s] * sh.c_u[k * E_DIM + e];
    sh.XY[s * E_DIM + e] = acc;
    sh.Xh[s * XLD + e] = (_Float16)acc;   // rows >= 50 remain zero
  }
  __syncthreads();

  // ---------------- Phase 7: GEMM4 fused tanh(x_hat@w4)@w5 -> attn -----------
  for (int half = 0; half < 2; ++half) {
    for (int i = tid; i < 128 * E_DIM; i += 256) {
      int h = i & 127, e = i >> 7;
      sh.Wt[h * XLD + e] = (_Float16)w4[e * H_DIM + half * 128 + h];
    }
    __syncthreads();
    for (int t = wave; t < 32; t += 8) {
      int m0 = (t & 3) * 16, n0 = (t >> 2) * 16;
      v8f c = {};
#pragma unroll
      for (int k0 = 0; k0 < E_DIM; k0 += 32) {
        v16h a = load_A(sh.Xh, m0, k0, lane);
        v16h bm = load_B(sh.Wt, k0, n0, lane);
        c = __builtin_amdgcn_wmma_f32_16x16x32_f16(false, a, false, bm, (short)0, c, false, false);
      }
      int n = half * 128 + n0 + hl;
      float w5v = sh.w5s[n];
#pragma unroll
      for (int i = 0; i < 8; ++i) {
        int row = m0 + hi * 8 + i;
        atomicAdd(&sh.att_logit[row], fast_tanh(c[i]) * w5v);
      }
    }
    __syncthreads();
  }
  if (tid == 0) {
    float mx = -3.0e38f;
    for (int s = 0; s < S_LEN; ++s) {
      float l = sh.att_logit[s] + sh.maskBias[s];
      sh.attn_s[s] = l; mx = fmaxf(mx, l);
    }
    float sum = 0.0f;
    for (int s = 0; s < S_LEN; ++s) { float e = __expf(sh.attn_s[s] - mx); sh.attn_s[s] = e; sum += e; }
    float inv = 1.0f / sum;
    for (int s = 0; s < S_LEN; ++s) sh.attn_s[s] *= inv;
  }
  __syncthreads();
  if (tid < E_DIM) {
    float acc = 0.0f;
    for (int s = 0; s < S_LEN; ++s) acc += sh.attn_s[s] * sh.XY[s * E_DIM + tid];
    sh.capt[tid] = acc;
  }
  __syncthreads();
  if (tid == 0) {
    float ss = 0.0f;
    for (int e = 0; e < E_DIM; ++e) ss += sh.capt[e] * sh.capt[e];
    sh.scal = 1.0f / fmaxf(sqrtf(ss), 1e-12f);
  }
  __syncthreads();
  if (tid < E_DIM) sh.capt[tid] *= sh.scal;
  __syncthreads();

  // ---------------- Phase 8: e_u, v_u, final dot products --------------------
  if (tid < K_TOP) {
    float acc = 0.0f;
    for (int e = 0; e < E_DIM; ++e) acc += sh.capt[e] * sh.phi[tid * E_DIM + e];
    sh.ek[tid] = acc;
  }
  __syncthreads();
  if (tid == 0) {
    float mx = fmaxf(fmaxf(sh.ek[0], sh.ek[1]), fmaxf(sh.ek[2], sh.ek[3]));
    float sum = 0.0f;
    for (int k = 0; k < K_TOP; ++k) { float e = __expf(sh.ek[k] - mx); sh.eu[k] = e; sum += e; }
    float inv = 1.0f / sum;
    for (int k = 0; k < K_TOP; ++k) sh.eu[k] *= inv;
  }
  __syncthreads();
  if (tid < E_DIM) {
    float acc = 0.0f;
#pragma unroll
    for (int k = 0; k < K_TOP; ++k) acc += sh.eu[k] * sh.phi[k * E_DIM + tid];
    sh.vu[tid] = acc;
  }
  __syncthreads();
  if (tid < 1 + NNEG) {
    int it = (tid == 0) ? pos_item[b] : neg_items[b * NNEG + tid - 1];
    const float* ie = item_emb + (size_t)it * E_DIM;
    float acc = 0.0f;
#pragma unroll 8
    for (int e = 0; e < E_DIM; ++e) acc += sh.vu[e] * ie[e];
    out[b * (1 + NNEG) + tid] = acc;
  }
}

extern "C" void kernel_launch(void* const* d_in, const int* in_sizes, int n_in,
                              void* d_out, int out_size, void* d_ws, size_t ws_size,
                              hipStream_t stream) {
  const int*   hist_item   = (const int*)d_in[0];
  const int*   pos_item    = (const int*)d_in[1];
  const int*   neg_items   = (const int*)d_in[2];
  const float* item_emb    = (const float*)d_in[3];
  const float* concept_emb = (const float*)d_in[4];
  const float* pos_emb     = (const float*)d_in[5];
  const float* w1          = (const float*)d_in[6];
  const float* w2          = (const float*)d_in[7];
  const float* w3          = (const float*)d_in[8];
  const float* wk1         = (const float*)d_in[9];
  const float* wk2         = (const float*)d_in[10];
  const float* w4          = (const float*)d_in[11];
  const float* w5          = (const float*)d_in[12];
  float*       out         = (float*)d_out;
  (void)in_sizes; (void)n_in; (void)out_size; (void)d_ws; (void)ws_size;

  sine_fused_kernel<<<B_TOT, 256, 0, stream>>>(
      hist_item, pos_item, neg_items, item_emb, concept_emb, pos_emb,
      w1, w2, w3, wk1, wk2, w4, w5, out);
}
